// LovaszLoss_32950989095302
// MI455X (gfx1250) — compile-verified, matrix operand order checked by reading, not run on hardware
//
#include <hip/hip_runtime.h>
#include <hip/hip_bf16.h>

typedef unsigned int u32;
typedef unsigned long long u64;
typedef __attribute__((ext_vector_type(8))) int v8i;
typedef __attribute__((ext_vector_type(4))) int v4i;
typedef __attribute__((ext_vector_type(4))) unsigned int v4u;

#define BIMG 16
#define PIX  (1u<<20)          // pixels per image
#define NB   256               // radix blocks per image (4096 elems each)
#define NBY  128               // label-sum blocks per image (8192 elems each)

#if defined(__has_builtin)
#  if __has_builtin(__builtin_amdgcn_tensor_load_to_lds) && __has_builtin(__builtin_amdgcn_s_wait_tensorcnt)
#    define HAVE_TDM 1
#  endif
#endif

// ---- key transform: monotone map so ascending u32 order == descending error ----
__device__ __forceinline__ u32 encodeKey(float e, u32 y) {
    u32 u   = __float_as_uint(e);
    u32 asc = (u & 0x80000000u) ? ~u : (u | 0x80000000u); // monotone increasing in e
    return ((~asc) & ~1u) | y;                            // descending key, label in LSB
}
__device__ __forceinline__ float decodeErr(u32 key) {
    u32 asc = ~key;
    u32 u   = (asc & 0x80000000u) ? (asc ^ 0x80000000u) : ~asc;
    return __uint_as_float(u);
}

// ---------------- K0: build keys ----------------
__global__ void k_init(const float* __restrict__ x, const int* __restrict__ lab,
                       u32* __restrict__ keys) {
    u32 gid = blockIdx.x * 256u + threadIdx.x;            // 4096 blocks -> 1M threads
#pragma unroll
    for (int k = 0; k < 16; ++k) {
        u32 i = gid + (u32)k * (4096u * 256u);
        float xi = x[i];
        u32 y = (u32)lab[i] & 1u;
        float e = 1.0f - xi * (y ? 1.0f : -1.0f);
        keys[i] = encodeKey(e, y);
    }
}

// ---------------- K1: per-block 16-digit histogram ----------------
__global__ void k_hist(const u32* __restrict__ keys, u32* __restrict__ hist, int shift) {
    __shared__ u32 cnt[16 * 256];
    const int t = threadIdx.x;
    const int blk = blockIdx.x & (NB - 1);
    const int img = blockIdx.x >> 8;
#pragma unroll
    for (int d = 0; d < 16; ++d) cnt[d * 256 + t] = 0;
    __syncthreads();
    const u32 base = ((u32)img << 20) + (u32)blk * 4096u + (u32)t * 16u;
#pragma unroll
    for (int j = 0; j < 16; ++j) {
        u32 d = (keys[base + j] >> shift) & 15u;
        cnt[d * 256 + t]++;                               // unique (d,t) writer: no atomics
    }
    __syncthreads();
    if (t < 16) {
        u32 s = 0;
        for (int i = 0; i < 256; ++i) s += cnt[t * 256 + i];
        hist[(((u32)img * 16u) + (u32)t) * (u32)NB + (u32)blk] = s;
    }
}

// ---------------- K2: per-image exclusive scan over [digit][block] (4096) ----------------
__global__ void k_scan(u32* __restrict__ hist) {
    __shared__ u32 tsum[256];
    const int t = threadIdx.x;
    u32* h = hist + (u32)blockIdx.x * 4096u;
    u32 loc[16]; u32 run = 0;
#pragma unroll
    for (int j = 0; j < 16; ++j) loc[j] = h[t * 16 + j];
#pragma unroll
    for (int j = 0; j < 16; ++j) { u32 v = loc[j]; loc[j] = run; run += v; }
    tsum[t] = run; __syncthreads();
    for (u32 off = 1; off < 256; off <<= 1) {
        u32 v = tsum[t];
        u32 a = (t >= (int)off) ? tsum[t - off] : 0u;
        __syncthreads();
        tsum[t] = v + a;
        __syncthreads();
    }
    u32 base = (t == 0) ? 0u : tsum[t - 1];
#pragma unroll
    for (int j = 0; j < 16; ++j) h[t * 16 + j] = base + loc[j];
}

// ---------------- K3: stable counting scatter ----------------
__global__ void k_scatter(const u32* __restrict__ src, u32* __restrict__ dst,
                          const u32* __restrict__ offs, int shift) {
    __shared__ u32 arr[16 * 256];
    __shared__ u32 tsum[256];
    __shared__ u32 gOff[16];
    __shared__ u32 dBase[16];
    const int t = threadIdx.x;
    const int blk = blockIdx.x & (NB - 1);
    const int img = blockIdx.x >> 8;
#pragma unroll
    for (int d = 0; d < 16; ++d) arr[d * 256 + t] = 0;
    __syncthreads();
    const u32 base = ((u32)img << 20) + (u32)blk * 4096u + (u32)t * 16u;
    u32 kb[16];
#pragma unroll
    for (int j = 0; j < 16; ++j) {
        kb[j] = src[base + j];
        u32 d = (kb[j] >> shift) & 15u;
        arr[d * 256 + t]++;
    }
    __syncthreads();
    // exclusive scan of arr[4096] in digit-major linear order
    u32 loc[16]; u32 run = 0;
#pragma unroll
    for (int j = 0; j < 16; ++j) loc[j] = arr[t * 16 + j];
#pragma unroll
    for (int j = 0; j < 16; ++j) { u32 v = loc[j]; loc[j] = run; run += v; }
    tsum[t] = run; __syncthreads();
    for (u32 off = 1; off < 256; off <<= 1) {
        u32 v = tsum[t];
        u32 a = (t >= (int)off) ? tsum[t - off] : 0u;
        __syncthreads();
        tsum[t] = v + a;
        __syncthreads();
    }
    u32 sbase = (t == 0) ? 0u : tsum[t - 1];
#pragma unroll
    for (int j = 0; j < 16; ++j) arr[t * 16 + j] = sbase + loc[j];
    __syncthreads();
    if (t < 16) {
        gOff[t]  = offs[(((u32)img * 16u) + (u32)t) * (u32)NB + (u32)blk];
        dBase[t] = arr[t * 256];                          // per-block base of digit t
    }
    __syncthreads();
    u64 seenLo = 0, seenHi = 0;                            // 16 packed 8-bit counters
#pragma unroll
    for (int j = 0; j < 16; ++j) {
        u32 key = kb[j];
        u32 d = (key >> shift) & 15u;
        u32 r;
        if (d < 8u) { r = (u32)((seenLo >> (8u * d)) & 0xFFu); seenLo += 1ull << (8u * d); }
        else { u32 dd = d - 8u; r = (u32)((seenHi >> (8u * dd)) & 0xFFu); seenHi += 1ull << (8u * dd); }
        u32 within = arr[d * 256 + t] - dBase[d];          // same digit, earlier threads
        u32 pos = gOff[d] + within + r;
        dst[((u32)img << 20) + pos] = key;
    }
}

// ---------------- K4: per-block label sums via V_WMMA_I32_16X16X64_IU8 ----------------
// B = all-ones bytes => every column total of C equals the grand total of A's 1024 bytes,
// independent of lane->(M,K) mapping. One WMMA sums 1024 labels per wave.
__global__ void k_ysum(const u32* __restrict__ keys, u32* __restrict__ bsY) {
    __shared__ u32 wsum[8];
    const int t = threadIdx.x, lane = t & 31, w = t >> 5;
    const int blk = blockIdx.x & (NBY - 1);
    const int img = blockIdx.x >> 7;
    const u32 seg = ((u32)img << 20) + (u32)blk * 8192u + (u32)w * 1024u;
    u32 pk[8];
#pragma unroll
    for (int q = 0; q < 8; ++q) pk[q] = 0;
#pragma unroll
    for (int j = 0; j < 32; ++j) {                         // coalesced across lanes
        u32 key = keys[seg + (u32)j * 32u + (u32)lane];
        pk[j >> 2] |= (key & 1u) << ((j & 3) * 8);
    }
    v8i a, bm;
    v8i c = {0, 0, 0, 0, 0, 0, 0, 0};
#pragma unroll
    for (int q = 0; q < 8; ++q) { a[q] = (int)pk[q]; bm[q] = 0x01010101; }
    c = __builtin_amdgcn_wmma_i32_16x16x64_iu8(false, a, false, bm, c, false, false);
    int partial = 0;
#pragma unroll
    for (int q = 0; q < 8; ++q) partial += c[q];           // sum over my M-half rows
    int tot = __shfl(partial, 0, 32) + __shfl(partial, 16, 32); // column-0 total
    if (lane == 0) wsum[w] = (u32)tot;
    __syncthreads();
    if (t == 0) {
        u32 s = 0;
        for (int i = 0; i < 8; ++i) s += wsum[i];
        bsY[blockIdx.x] = s;
    }
}

// ---------------- K5: per-image exclusive scan of 128 block sums ----------------
__global__ void k_yscan(const u32* __restrict__ bsY, u32* __restrict__ baseCP,
                        u32* __restrict__ Pimg) {
    __shared__ u32 s[NBY];
    const int t = threadIdx.x, img = blockIdx.x;
    s[t] = bsY[img * NBY + t]; __syncthreads();
    for (u32 off = 1; off < NBY; off <<= 1) {
        u32 v = s[t];
        u32 a = (t >= (int)off) ? s[t - off] : 0u;
        __syncthreads();
        s[t] = v + a;
        __syncthreads();
    }
    baseCP[img * NBY + t] = (t == 0) ? 0u : s[t - 1];
    if (t == NBY - 1) Pimg[img] = s[NBY - 1];
}

// ---------------- K6: Lovasz gradient + dot, per-block partials ----------------
// Tile of 8192 sorted keys is staged into LDS by the Tensor Data Mover (one DMA
// issued by wave 0), using TDM padding (1 dword per 32) so the per-thread
// sequential chunk tile[t*33+j] is bank-conflict-free (gcd(33,64)=1).
__global__ void k_loss(const u32* __restrict__ keys, const u32* __restrict__ baseCP,
                       const u32* __restrict__ Pimg, float* __restrict__ lossPart) {
    __shared__ u32 tile[8192 + 256];                       // 8192 keys + pad dwords
    __shared__ u32 tsum[256];
    __shared__ float red[256];
    const int t = threadIdx.x;
    const int blk = blockIdx.x & (NBY - 1);
    const int img = blockIdx.x >> 7;
    const u32 blockBase = ((u32)img << 20) + (u32)blk * 8192u;

#if defined(HAVE_TDM)
    if (t < 32) {                                          // wave 0 issues the DMA
        __attribute__((address_space(3))) u32* lp =
            (__attribute__((address_space(3))) u32*)&tile[0];
        u32 ldsAddr = (u32)(size_t)lp;
        u64 gaddr = (u64)(size_t)(keys + blockBase);
        v4u g0; v8i g1; v4i g2; v4i g3; v8i g4;
        g0[0] = 1u;                                        // count=1 (valid descriptor)
        g0[1] = ldsAddr;                                   // LDS dest byte address
        g0[2] = (u32)(gaddr & 0xFFFFFFFFull);              // global_addr[31:0]
        g0[3] = (u32)((gaddr >> 32) & 0x01FFFFFFull) | (2u << 30); // addr[56:32], type=2
        // data_size=4B (2), pad_enable=1, pad_interval=32 dwords (4), pad_amount=1 (0)
        g1[0] = (int)(0x00020000u | 0x00100000u | (4u << 22));
        g1[1] = (int)((8192u & 0xFFFFu) << 16);            // tensor_dim0 = 8192 (lo16)
        g1[2] = (int)(1u << 16);                           // dim0 hi=0 | tensor_dim1=1
        g1[3] = (int)((8192u & 0xFFFFu) << 16);            // tile_dim0 = 8192
        g1[4] = 1;                                         // tile_dim1 = 1
        g1[5] = 8192;                                      // tensor_dim0_stride (lo32)
        g1[6] = 0; g1[7] = 0;
        g2[0] = 0; g2[1] = 0; g2[2] = 0; g2[3] = 0;
        g3[0] = 0; g3[1] = 0; g3[2] = 0; g3[3] = 0;
        g4[0] = 0; g4[1] = 0; g4[2] = 0; g4[3] = 0;
        g4[4] = 0; g4[5] = 0; g4[6] = 0; g4[7] = 0;
        // clang-23 / amdgpu-toolchain arity: 6 args (extra int32x8 group, then cpol)
        __builtin_amdgcn_tensor_load_to_lds(g0, g1, g2, g3, g4, 0);
        __builtin_amdgcn_s_wait_tensorcnt((unsigned short)0);
    }
    __syncthreads();
#else
    for (int j = 0; j < 32; ++j) {                         // coalesced fallback staging
        u32 i = (u32)j * 256u + (u32)t;
        tile[i + i / 32u] = keys[blockBase + i];           // same padded layout
    }
    __syncthreads();
#endif

    // phase 1: per-thread label sum over its sequential chunk (padded LDS reads)
    u32 ys = 0;
#pragma unroll
    for (int j = 0; j < 32; ++j) ys += tile[t * 33 + j] & 1u;
    tsum[t] = ys; __syncthreads();
    for (u32 off = 1; off < 256; off <<= 1) {
        u32 v = tsum[t];
        u32 a = (t >= (int)off) ? tsum[t - off] : 0u;
        __syncthreads();
        tsum[t] = v + a;
        __syncthreads();
    }
    u32 running = baseCP[img * NBY + blk] + ((t == 0) ? 0u : tsum[t - 1]);
    const float Pf = (float)Pimg[img];
    const u32 rank0 = (u32)blk * 8192u + (u32)t * 32u;
    float acc = 0.0f;
#pragma unroll
    for (int j = 0; j < 32; ++j) {
        u32 key = tile[t * 33 + j];
        u32 y = key & 1u;
        running += y;                                      // inclusive cum positives
        float cp = (float)running;
        float k1 = (float)(rank0 + (u32)j + 1u);           // 1-indexed rank
        float jac = 1.0f - (Pf - cp) / (Pf + k1 - cp);
        float jprev = 0.0f;
        if (rank0 + (u32)j > 0u) {
            float cpm = cp - (float)y;
            jprev = 1.0f - (Pf - cpm) / (Pf + (k1 - 1.0f) - cpm);
        }
        float e = decodeErr(key);
        acc += fmaxf(e, 0.0f) * (jac - jprev);
    }
    red[t] = acc; __syncthreads();
    for (int s2 = 128; s2 > 0; s2 >>= 1) {                 // fixed-order tree
        if (t < s2) red[t] += red[t + s2];
        __syncthreads();
    }
    if (t == 0) lossPart[blockIdx.x] = red[0];
}

// ---------------- K7: deterministic final mean ----------------
__global__ void k_final(const float* __restrict__ lossPart, float* __restrict__ out) {
    if (threadIdx.x == 0 && blockIdx.x == 0) {
        float tot = 0.0f;
        for (int img = 0; img < BIMG; ++img) {
            float s = 0.0f;
            for (int b = 0; b < NBY; ++b) s += lossPart[img * NBY + b];
            tot += s;
        }
        out[0] = tot / (float)BIMG;
    }
}

extern "C" void kernel_launch(void* const* d_in, const int* in_sizes, int n_in,
                              void* d_out, int out_size, void* d_ws, size_t ws_size,
                              hipStream_t stream) {
    const float* x   = (const float*)d_in[0];
    const int*   lab = (const int*)d_in[1];
    float* out = (float*)d_out;
    (void)in_sizes; (void)n_in; (void)out_size; (void)ws_size;

    unsigned char* w = (unsigned char*)d_ws;
    u32* bufA = (u32*)w;                                   // 64 MB keys
    u32* bufB = (u32*)(w + ((size_t)64 << 20));            // 64 MB keys
    u32* hist = (u32*)(w + ((size_t)128 << 20));           // 16*16*256 u32
    u32* bsY    = hist + (size_t)BIMG * 16 * NB;           // 16*128
    u32* baseCP = bsY  + (size_t)BIMG * NBY;               // 16*128
    u32* Pimg   = baseCP + (size_t)BIMG * NBY;             // 16
    float* lossPart = (float*)(Pimg + BIMG);               // 16*128

    k_init<<<4096, 256, 0, stream>>>(x, lab, bufA);

    u32* src = bufA; u32* dst = bufB;
    for (int p = 0; p < 8; ++p) {                          // 8 x 4-bit LSD passes
        const int shift = p * 4;
        k_hist<<<BIMG * NB, 256, 0, stream>>>(src, hist, shift);
        k_scan<<<BIMG, 256, 0, stream>>>(hist);
        k_scatter<<<BIMG * NB, 256, 0, stream>>>(src, dst, hist, shift);
        u32* tmp = src; src = dst; dst = tmp;
    }
    // after an even number of passes, sorted keys are back in bufA (== src)
    k_ysum<<<BIMG * NBY, 256, 0, stream>>>(src, bsY);
    k_yscan<<<BIMG, NBY, 0, stream>>>(bsY, baseCP, Pimg);
    k_loss<<<BIMG * NBY, 256, 0, stream>>>(src, baseCP, Pimg, lossPart);
    k_final<<<1, 32, 0, stream>>>(lossPart, out);
}
